// Informer_11287174054238
// MI455X (gfx1250) — matrix operand order repaired and checked by reading, not compiled, not run on hardware
//
#include <hip/hip_runtime.h>
#include <hip/hip_bf16.h>
#include <math.h>

#define BB 8
#define LL 2048
#define DD 256
#define CIN 32
#define SK 40
#define UU 40
#define BL (BB*LL)
#define G3 (3*DD)
#define KT 8                      // K=256 in steps of 32 for every GEMM here

typedef _Float16 h8   __attribute__((ext_vector_type(8)));
typedef _Float16 v16h __attribute__((ext_vector_type(16)));
typedef float    v8f  __attribute__((ext_vector_type(8)));
typedef unsigned int u32x4 __attribute__((ext_vector_type(4)));
typedef int          i32x4 __attribute__((ext_vector_type(4)));
typedef int          i32x8 __attribute__((ext_vector_type(8)));

union HV { struct { h8 lo, hi; } p; v16h v; };

__device__ inline float wave_sum(float v){ for(int o=16;o;o>>=1) v += __shfl_xor(v,o,32); return v; }
__device__ inline float wave_max(float v){ for(int o=16;o;o>>=1) v = fmaxf(v,__shfl_xor(v,o,32)); return v; }

__device__ inline float dot256h(const _Float16* __restrict__ a, const _Float16* __restrict__ b){
  float s = 0.f;
  #pragma unroll 4
  for (int i = 0; i < DD; i += 8) {
    h8 av = *(const h8*)(a + i);
    h8 bv = *(const h8*)(b + i);
    #pragma unroll
    for (int j = 0; j < 8; ++j) s += (float)av[j] * (float)bv[j];
  }
  return s;
}

// ---- Tensor Data Mover: 1-D contiguous global->LDS copy (bytes % 8 == 0) ----
// D# group0: count=1 (valid), lds_addr, global_addr, type=2
// D# group1: data_size=3 (8B), tensor_dim0=elems, tensor_dim1=1, tile_dim0=elems,
//            tensor_dim0_stride=elems
__device__ inline void tdm_load_1d(unsigned lds_addr, const void* gptr, unsigned bytes)
{
  unsigned long long ga = (unsigned long long)(uintptr_t)gptr;
  unsigned elems = bytes >> 3;
  u32x4 g0;
  g0[0] = 1u;                                                // count=1
  g0[1] = lds_addr;                                          // lds_addr [63:32]
  g0[2] = (unsigned)(ga & 0xffffffffu);                      // global_addr lo
  g0[3] = (unsigned)((ga >> 32) & 0x1ffffffu) | (2u << 30);  // global_addr hi | type=2
  i32x8 g1;
  g1[0] = (int)(3u << 16);                                   // data_size=3 (8 bytes)
  g1[1] = (int)((elems & 0xffffu) << 16);                    // tensor_dim0[15:0]  @ bits 63:48
  g1[2] = (int)(((elems >> 16) & 0xffffu) | (1u << 16));     // tensor_dim0 hi | tensor_dim1=1
  g1[3] = (int)((elems & 0xffffu) << 16);                    // tile_dim0 @ bits 127:112
  g1[4] = 0;                                                 // tile_dim1/2 unused
  g1[5] = (int)elems;                                        // tensor_dim0_stride lo
  g1[6] = 0;
  g1[7] = 0;
  i32x4 z4 = {0,0,0,0};
#if defined(__clang_major__) && (__clang_major__ >= 23)
  i32x8 z8 = {0,0,0,0,0,0,0,0};
  __builtin_amdgcn_tensor_load_to_lds(g0, g1, z4, z4, z8, 0);
#else
  __builtin_amdgcn_tensor_load_to_lds(g0, g1, z4, z4, 0);
#endif
}

// ---------------- weight packing for WMMA B-operand ----------------
// lane L of tile (kt,nt) holds column n = nt*16 + L%16, K = kt*32 + (L<16?0:16) + i
// stored contiguously: Wp[((nt*KT + kt)*32 + L)*16 + i]  -> an N-group of 4 tiles
// spans a contiguous 32KB region, ideal for one TDM descriptor.
__global__ __launch_bounds__(256)
void pack_w_k(const float* __restrict__ W, _Float16* __restrict__ Wp,
              int K, int N, int transposed)
{
  int t = blockIdx.x*256 + threadIdx.x;
  int lane = t & 31;
  int kt = (t >> 5) % KT;
  int nt = (t >> 5) / KT;
  if (nt >= (N >> 4)) return;
  int n = nt*16 + (lane & 15);
  int kbase = kt*32 + ((lane >> 4) << 4);
  _Float16* dst = Wp + ((size_t)((nt*KT + kt)*32 + lane))*16;
  #pragma unroll
  for (int i = 0; i < 16; ++i) {
    int k = kbase + i;
    float v = transposed ? W[(size_t)n*K + k] : W[(size_t)k*N + n];
    dst[i] = (_Float16)v;
  }
}

// ---------------- WMMA GEMM: out = act(A[MxK] @ W[KxN] + bias), K=256 ----------------
// Block: 8 waves = 8 M-tiles x one 64-wide N-group. B strip staged in LDS via TDM.
__global__ __launch_bounds__(256)
void gemm_wmma_k(const _Float16* __restrict__ A, const _Float16* __restrict__ Wp,
                 const float* __restrict__ bias, float* __restrict__ outF,
                 _Float16* __restrict__ outH, int M, int N, int act)
{
  constexpr int K = 256;
  __shared__ __align__(16) _Float16 Bsh[4*KT*32*16];   // 32KB: 4 N-tiles x full K
  int ngroups = N >> 6;
  int mb = blockIdx.x / ngroups;
  int ng = blockIdx.x - mb*ngroups;
  int wave = threadIdx.x >> 5;
  int lane = threadIdx.x & 31;

  if (wave == 0) {
    tdm_load_1d((unsigned)(uintptr_t)Bsh,
                Wp + (size_t)(ng*4)*KT*32*16,
                (unsigned)(sizeof(_Float16)*4*KT*32*16));
    __builtin_amdgcn_s_wait_tensorcnt(0);
  }
  __syncthreads();

  int mt = mb*8 + wave;                       // M is a multiple of 128 in all calls
  v8f acc[4] = {};
  int mrow = mt*16 + (lane & 15);
  const _Float16* arow = A + (size_t)mrow*K + ((lane >> 4) << 4);
  const _Float16* bl = Bsh + lane*16;
  #pragma unroll 2
  for (int kt = 0; kt < KT; ++kt) {
    HV a;
    a.p.lo = *(const h8*)(arow);
    a.p.hi = *(const h8*)(arow + 8);
    arow += 32;
    #pragma unroll
    for (int j = 0; j < 4; ++j) {
      const _Float16* bp = bl + (size_t)(j*KT + kt)*32*16;
      HV b;
      b.p.lo = *(const h8*)bp;
      b.p.hi = *(const h8*)(bp + 8);
      acc[j] = __builtin_amdgcn_wmma_f32_16x16x32_f16(
          false, a.v, false, b.v, (short)0, acc[j], false, false);
    }
  }
  int mr0 = mt*16 + ((lane >> 4) << 3);
  #pragma unroll
  for (int j = 0; j < 4; ++j) {
    int n = ng*64 + j*16 + (lane & 15);
    float bs = bias ? bias[n] : 0.0f;
    #pragma unroll
    for (int r = 0; r < 8; ++r) {
      float v = acc[j][r] + bs;
      if (act == 1) v = 0.5f*v*(1.0f + erff(v*0.70710678118f));   // exact GELU
      size_t off = (size_t)(mr0 + r)*N + n;
      if (outF) outF[off] = v;
      if (outH) outH[off] = (_Float16)v;
    }
  }
}

// ---------------- embedding + positional encoding ----------------
__global__ __launch_bounds__(256)
void embed_k(const float* __restrict__ x, const float* __restrict__ emb_w,
             const float* __restrict__ emb_b, float* __restrict__ srcf,
             _Float16* __restrict__ srch)
{
  int t = blockIdx.x*256 + threadIdx.x;
  if (t >= BL*DD) return;
  int d = t & (DD-1);
  int row = t >> 8;
  int l = row & (LL-1);
  const float* xr = x + (size_t)row*CIN;
  float s = emb_b[d];
  #pragma unroll
  for (int c = 0; c < CIN; ++c) s += xr[c]*emb_w[c*DD + d];
  float div = expf(-(float)(d & ~1) * (9.21034037198f/(float)DD));
  float arg = (float)l * div;
  s += (d & 1) ? cosf(arg) : sinf(arg);
  srcf[t] = s;
  srch[t] = (_Float16)s;
}

// ---------------- layer norm (wave per 256-wide row) ----------------
__global__ __launch_bounds__(256)
void ln_k(const float* __restrict__ in1, const float* __restrict__ in2,
          const float* __restrict__ g, const float* __restrict__ b,
          float* __restrict__ outF, _Float16* __restrict__ outH, int rows)
{
  int wid  = (blockIdx.x*blockDim.x + threadIdx.x) >> 5;
  int lane = threadIdx.x & 31;
  if (wid >= rows) return;
  const float* p1 = in1 + (size_t)wid*DD;
  const float* p2 = in2 ? in2 + (size_t)wid*DD : nullptr;
  float v[8];
  #pragma unroll
  for (int i = 0; i < 8; ++i) {
    float t = p1[lane + 32*i];
    if (p2) t += p2[lane + 32*i];
    v[i] = t;
  }
  float s = 0.f;
  #pragma unroll
  for (int i = 0; i < 8; ++i) s += v[i];
  float mu = wave_sum(s) * (1.0f/DD);
  float var = 0.f;
  #pragma unroll
  for (int i = 0; i < 8; ++i) { float d0 = v[i]-mu; var += d0*d0; }
  var = wave_sum(var) * (1.0f/DD);
  float inv = rsqrtf(var + 1e-5f);
  #pragma unroll
  for (int i = 0; i < 8; ++i) {
    int d = lane + 32*i;
    float o = (v[i]-mu)*inv*g[d] + b[d];
    size_t off = (size_t)wid*DD + d;
    if (outF) outF[off] = o;
    if (outH) outH[off] = (_Float16)o;
  }
}

// ---------------- sparse sampling score M[b,l] ----------------
__global__ __launch_bounds__(256)
void probe_k(const _Float16* __restrict__ Qh, const _Float16* __restrict__ Kh,
             const int* __restrict__ sidx, float* __restrict__ Mout)
{
  int wid  = (blockIdx.x*blockDim.x + threadIdx.x) >> 5;
  int lane = threadIdx.x & 31;
  if (wid >= BL) return;
  int b = wid >> 11;
  int l = wid & (LL-1);
  const _Float16* q  = Qh + (size_t)wid*DD;
  const _Float16* kb = Kh + (size_t)b*LL*DD;
  float d0 = dot256h(q, kb + (size_t)sidx[l*SK + lane]*DD);
  float lm = d0, ls = d0;
  if (lane < SK - 32) {
    float d1 = dot256h(q, kb + (size_t)sidx[l*SK + 32 + lane]*DD);
    lm = fmaxf(lm, d1); ls += d1;
  }
  float mx = wave_max(lm);
  float sm = wave_sum(ls);
  if (lane == 0) Mout[wid] = mx - sm*(1.0f/(float)LL);
}

// ---------------- top-k (iterative argmax, tie -> smaller index) ----------------
__global__ __launch_bounds__(256)
void topk_k(const float* __restrict__ Mbuf, int* __restrict__ top)
{
  int b = blockIdx.x, tid = threadIdx.x;
  __shared__ float sm[LL];
  __shared__ float rv[256];
  __shared__ int   ri[256];
  for (int i = tid; i < LL; i += 256) sm[i] = Mbuf[(size_t)b*LL + i];
  __syncthreads();
  for (int it = 0; it < UU; ++it) {
    float bv = -3.4e38f; int bi = 0;
    #pragma unroll
    for (int j = 0; j < LL/256; ++j) {
      int i = tid + 256*j;
      float v = sm[i];
      if (v > bv) { bv = v; bi = i; }
    }
    rv[tid] = bv; ri[tid] = bi;
    __syncthreads();
    for (int s = 128; s; s >>= 1) {
      if (tid < s) {
        float v2 = rv[tid+s]; int i2 = ri[tid+s];
        if (v2 > rv[tid] || (v2 == rv[tid] && i2 < ri[tid])) { rv[tid]=v2; ri[tid]=i2; }
      }
      __syncthreads();
    }
    if (tid == 0) { top[b*UU + it] = ri[0]; sm[ri[0]] = -3.4e38f; }
    __syncthreads();
  }
}

// ---------------- dense scores for selected queries ----------------
__global__ __launch_bounds__(256)
void scores_k(const _Float16* __restrict__ Qh, const _Float16* __restrict__ Kh,
              const int* __restrict__ top, float* __restrict__ attn)
{
  int wid  = (blockIdx.x*blockDim.x + threadIdx.x) >> 5;
  int lane = threadIdx.x & 31;
  if (wid >= BB*UU*(LL/32)) return;
  int kb = wid % (LL/32);
  int u  = (wid / (LL/32)) % UU;
  int b  = wid / (LL/32) / UU;
  int qrow = top[b*UU + u];
  const _Float16* q = Qh + ((size_t)b*LL + qrow)*DD;
  const _Float16* k = Kh + ((size_t)b*LL + kb*32 + lane)*DD;
  float s = dot256h(q, k);
  attn[((size_t)(b*UU + u))*LL + kb*32 + lane] = s * 0.0625f;   // /sqrt(256)
}

__global__ __launch_bounds__(256)
void softmax_k(float* __restrict__ attn)
{
  int row = blockIdx.x, tid = threadIdx.x;
  float* p = attn + (size_t)row*LL;
  __shared__ float red[256];
  float mx = -3.4e38f;
  for (int i = tid; i < LL; i += 256) mx = fmaxf(mx, p[i]);
  red[tid] = mx; __syncthreads();
  for (int s = 128; s; s >>= 1) { if (tid < s) red[tid] = fmaxf(red[tid], red[tid+s]); __syncthreads(); }
  mx = red[0]; __syncthreads();
  float sm = 0.f;
  for (int i = tid; i < LL; i += 256) { float e = expf(p[i]-mx); p[i] = e; sm += e; }
  red[tid] = sm; __syncthreads();
  for (int s = 128; s; s >>= 1) { if (tid < s) red[tid] += red[tid+s]; __syncthreads(); }
  float inv = 1.0f/red[0];
  for (int i = tid; i < LL; i += 256) p[i] *= inv;
}

__global__ __launch_bounds__(256)
void vmean_k(const _Float16* __restrict__ Vh, _Float16* __restrict__ vmeanH)
{
  int t = blockIdx.x*256 + threadIdx.x;
  if (t >= BB*DD) return;
  int d = t & (DD-1), b = t >> 8;
  const _Float16* vb = Vh + (size_t)b*LL*DD + d;
  float s = 0.f;
  for (int k = 0; k < LL; ++k) s += (float)vb[(size_t)k*DD];
  vmeanH[t] = (_Float16)(s*(1.0f/(float)LL));
}

__global__ __launch_bounds__(256)
void ctxfill_k(const _Float16* __restrict__ vmeanH, _Float16* __restrict__ ctxh)
{
  int t = blockIdx.x*256 + threadIdx.x;
  if (t >= BL*DD) return;
  int d = t & (DD-1);
  int b = (t >> 8) >> 11;
  ctxh[t] = vmeanH[b*DD + d];
}

__global__ __launch_bounds__(256)
void upd_scatter_k(const float* __restrict__ attn, const _Float16* __restrict__ Vh,
                   const int* __restrict__ top, _Float16* __restrict__ ctxh)
{
  int t = blockIdx.x*256 + threadIdx.x;
  if (t >= BB*UU*DD) return;
  int d = t & (DD-1);
  int u = (t >> 8) % UU;
  int b = (t >> 8) / UU;
  const float* arow = attn + ((size_t)(b*UU + u))*LL;
  const _Float16* vb = Vh + (size_t)b*LL*DD + d;
  float s = 0.f;
  for (int k = 0; k < LL; ++k) s += arow[k]*(float)vb[(size_t)k*DD];
  int l = top[b*UU + u];
  ctxh[((size_t)(b*LL + l))*DD + d] = (_Float16)s;
}

// ---------------- GRU pieces ----------------
__global__ __launch_bounds__(256)
void zero_h_k(float* __restrict__ hf, _Float16* __restrict__ hh)
{
  int t = blockIdx.x*256 + threadIdx.x;
  if (t < LL*DD) { hf[t] = 0.f; hh[t] = (_Float16)0.f; }
}

__global__ __launch_bounds__(256)
void gru_gate_k(const float* __restrict__ gi, const float* __restrict__ gh,
                float* __restrict__ hf, _Float16* __restrict__ hh,
                float* __restrict__ hlast, int bstep)
{
  int t = blockIdx.x*256 + threadIdx.x;
  if (t >= LL*DD) return;
  int d = t & (DD-1), l = t >> 8;
  const float* gir = gi + (size_t)l*G3;
  const float* ghr = gh + (size_t)l*G3;
  float r = 1.f/(1.f + expf(-(gir[d]       + ghr[d])));
  float z = 1.f/(1.f + expf(-(gir[DD+d]    + ghr[DD+d])));
  float n = tanhf(gir[2*DD+d] + r*ghr[2*DD+d]);
  float hold = hf[t];
  float h = (1.f - z)*n + z*hold;
  hf[t] = h;
  hh[t] = (_Float16)h;
  if (l == LL-1) hlast[bstep*DD + d] = h;
}

__global__ __launch_bounds__(256)
void fc_k(const float* __restrict__ hlast, const float* __restrict__ fw,
          const float* __restrict__ fb, float* __restrict__ out)
{
  int wave = threadIdx.x >> 5, lane = threadIdx.x & 31;
  float s = 0.f;
  #pragma unroll
  for (int i = 0; i < 8; ++i) { int d = lane + 32*i; s += hlast[wave*DD + d]*fw[d]; }
  s = wave_sum(s);
  if (lane == 0) out[wave] = s + fb[0];
}

// ---------------- host driver ----------------
static inline int gemm_blocks(int M, int N) {
  return (M >> 7) * (N >> 6);          // blocks of 8 M-tiles x 64-wide N-group
}

extern "C" void kernel_launch(void* const* d_in, const int* in_sizes, int n_in,
                              void* d_out, int out_size, void* d_ws, size_t ws_size,
                              hipStream_t stream)
{
  const float* x      = (const float*)d_in[0];
  const int*   sidx   = (const int*)  d_in[1];
  const float* emb_w  = (const float*)d_in[2];
  const float* emb_b  = (const float*)d_in[3];
  const float* wq = (const float*)d_in[4];  const float* bq = (const float*)d_in[5];
  const float* wk = (const float*)d_in[6];  const float* bk = (const float*)d_in[7];
  const float* wv = (const float*)d_in[8];  const float* bv = (const float*)d_in[9];
  const float* wo = (const float*)d_in[10]; const float* bo = (const float*)d_in[11];
  const float* c1w = (const float*)d_in[12]; const float* c1b = (const float*)d_in[13];
  const float* c2w = (const float*)d_in[14]; const float* c2b = (const float*)d_in[15];
  const float* ln1g = (const float*)d_in[16]; const float* ln1b = (const float*)d_in[17];
  const float* ln2g = (const float*)d_in[18]; const float* ln2b = (const float*)d_in[19];
  const float* lnfg = (const float*)d_in[20]; const float* lnfb = (const float*)d_in[21];
  const float* wih = (const float*)d_in[22]; const float* whh = (const float*)d_in[23];
  const float* bih = (const float*)d_in[24]; const float* bhh = (const float*)d_in[25];
  const float* fcw = (const float*)d_in[26]; const float* fcb = (const float*)d_in[27];
  float* out = (float*)d_out;

  char* ws = (char*)d_ws;
  size_t cur = 0;
  auto alloc = [&](size_t bytes) -> char* {
    char* p = ws + cur;
    cur += (bytes + 255) & ~(size_t)255;
    return p;
  };

  float*    srcf = (float*)   alloc((size_t)BL*DD*4);
  _Float16* srch = (_Float16*)alloc((size_t)BL*DD*2);
  _Float16* Qh   = (_Float16*)alloc((size_t)BL*DD*2);
  _Float16* Kh   = (_Float16*)alloc((size_t)BL*DD*2);
  _Float16* Vh   = (_Float16*)alloc((size_t)BL*DD*2);
  _Float16* ctxh = (_Float16*)alloc((size_t)BL*DD*2);
  float*    tmpf = (float*)   alloc((size_t)BL*DD*4);
  float*    x1f  = (float*)   alloc((size_t)BL*DD*4);
  _Float16* x1h  = (_Float16*)alloc((size_t)BL*DD*2);
  _Float16* yh   = (_Float16*)alloc((size_t)BL*DD*2);
  _Float16* ench = (_Float16*)alloc((size_t)BL*DD*2);
  float*    Mbuf = (float*)   alloc((size_t)BB*LL*4);
  int*      top  = (int*)     alloc((size_t)BB*UU*4);
  float*    attn = (float*)   alloc((size_t)BB*UU*LL*4);
  _Float16* vmH  = (_Float16*)alloc((size_t)BB*DD*2);
  float*    gif  = (float*)   alloc((size_t)LL*G3*4);
  float*    ghf  = (float*)   alloc((size_t)LL*G3*4);
  float*    hf   = (float*)   alloc((size_t)LL*DD*4);
  _Float16* hh   = (_Float16*)alloc((size_t)LL*DD*2);
  float*    hlast= (float*)   alloc((size_t)BB*DD*4);
  _Float16* pQ   = (_Float16*)alloc((size_t)DD*DD*2);
  _Float16* pK   = (_Float16*)alloc((size_t)DD*DD*2);
  _Float16* pV   = (_Float16*)alloc((size_t)DD*DD*2);
  _Float16* pO   = (_Float16*)alloc((size_t)DD*DD*2);
  _Float16* pC1  = (_Float16*)alloc((size_t)DD*DD*2);
  _Float16* pC2  = (_Float16*)alloc((size_t)DD*DD*2);
  _Float16* pIH  = (_Float16*)alloc((size_t)DD*G3*2);
  _Float16* pHH  = (_Float16*)alloc((size_t)DD*G3*2);
  (void)ws_size; (void)in_sizes; (void)n_in; (void)out_size;

  // ---- pack weights into WMMA-B lane layout ----
  int pk256 = ((DD/16)*KT*32 + 255)/256;
  int pk768 = ((G3/16)*KT*32 + 255)/256;
  pack_w_k<<<pk256,256,0,stream>>>(wq,  pQ,  DD, DD, 0);
  pack_w_k<<<pk256,256,0,stream>>>(wk,  pK,  DD, DD, 0);
  pack_w_k<<<pk256,256,0,stream>>>(wv,  pV,  DD, DD, 0);
  pack_w_k<<<pk256,256,0,stream>>>(wo,  pO,  DD, DD, 0);
  pack_w_k<<<pk256,256,0,stream>>>(c1w, pC1, DD, DD, 0);
  pack_w_k<<<pk256,256,0,stream>>>(c2w, pC2, DD, DD, 0);
  pack_w_k<<<pk768,256,0,stream>>>(wih, pIH, DD, G3, 1);   // gi = enc @ wih^T
  pack_w_k<<<pk768,256,0,stream>>>(whh, pHH, DD, G3, 1);   // gh = h   @ whh^T

  // ---- embedding + positional encoding ----
  embed_k<<<(BL*DD+255)/256,256,0,stream>>>(x, emb_w, emb_b, srcf, srch);

  // ---- Q, K, V projections (WMMA + TDM-staged weights) ----
  int gb = gemm_blocks(BL, DD);
  gemm_wmma_k<<<gb,256,0,stream>>>(srch, pQ, bq, nullptr, Qh, BL, DD, 0);
  gemm_wmma_k<<<gb,256,0,stream>>>(srch, pK, bk, nullptr, Kh, BL, DD, 0);
  gemm_wmma_k<<<gb,256,0,stream>>>(srch, pV, bv, nullptr, Vh, BL, DD, 0);

  // ---- ProbSparse attention ----
  probe_k<<<(BL*32+255)/256,256,0,stream>>>(Qh, Kh, sidx, Mbuf);
  topk_k<<<BB,256,0,stream>>>(Mbuf, top);
  scores_k<<<(BB*UU*(LL/32)*32+255)/256,256,0,stream>>>(Qh, Kh, top, attn);
  softmax_k<<<BB*UU,256,0,stream>>>(attn);
  vmean_k<<<(BB*DD+255)/256,256,0,stream>>>(Vh, vmH);
  ctxfill_k<<<(BL*DD+255)/256,256,0,stream>>>(vmH, ctxh);
  upd_scatter_k<<<(BB*UU*DD+255)/256,256,0,stream>>>(attn, Vh, top, ctxh);

  // ---- output projection + LN1 ----
  gemm_wmma_k<<<gb,256,0,stream>>>(ctxh, pO, bo, tmpf, nullptr, BL, DD, 0);
  ln_k<<<(BL*32+255)/256,256,0,stream>>>(srcf, tmpf, ln1g, ln1b, x1f, x1h, BL);

  // ---- FFN: gelu(x1@c1w+b) @ c2w + b, then LN2 then LNf ----
  gemm_wmma_k<<<gb,256,0,stream>>>(x1h, pC1, c1b, nullptr, yh, BL, DD, 1);
  gemm_wmma_k<<<gb,256,0,stream>>>(yh,  pC2, c2b, tmpf, nullptr, BL, DD, 0);
  ln_k<<<(BL*32+255)/256,256,0,stream>>>(x1f, tmpf, ln2g, ln2b, srcf, nullptr, BL);
  ln_k<<<(BL*32+255)/256,256,0,stream>>>(srcf, nullptr, lnfg, lnfb, nullptr, ench, BL);

  // ---- GRU scan over the 8 batch steps ----
  zero_h_k<<<(LL*DD+255)/256,256,0,stream>>>(hf, hh);
  int gbg = gemm_blocks(LL, G3);
  for (int b = 0; b < BB; ++b) {
    gemm_wmma_k<<<gbg,256,0,stream>>>(ench + (size_t)b*LL*DD, pIH, bih, gif, nullptr, LL, G3, 0);
    gemm_wmma_k<<<gbg,256,0,stream>>>(hh, pHH, bhh, ghf, nullptr, LL, G3, 0);
    gru_gate_k<<<(LL*DD+255)/256,256,0,stream>>>(gif, ghf, hf, hh, hlast, b);
  }

  // ---- final FC: (8,256) @ (256,1) + b ----
  fc_k<<<1,256,0,stream>>>(hlast, fcw, fcb, out);
}
